// HeteroGAE_Decoder_35579509080814
// MI455X (gfx1250) — matrix operand order repaired and checked by reading, not compiled, not run on hardware
//
#include <hip/hip_runtime.h>
#include <hip/hip_bf16.h>
#include <math.h>

#define N_NODES 200000
#define N_EDGES 1600000
#define N_PAIRS 1000000
#define DD      64
#define HH      256
#define CHN     128
#define XDIM    20
#define LAYERS  3

typedef __attribute__((ext_vector_type(16))) _Float16 v16h;
typedef __attribute__((ext_vector_type(8)))  float    v8f;
typedef __attribute__((ext_vector_type(4)))  unsigned int u32x4;
typedef __attribute__((ext_vector_type(8)))  int i32x8;
typedef __attribute__((ext_vector_type(4)))  int i32x4;

// Tensor Data Mover load: arity differs between toolchains (probe-confirmed).
#if defined(__clang_major__) && (__clang_major__ >= 23)
#define TDM_LOAD_TO_LDS(g0, g1, g2, g3) \
  __builtin_amdgcn_tensor_load_to_lds((g0), (g1), (g2), (g3), (i32x8){0,0,0,0,0,0,0,0}, 0)
#else
#define TDM_LOAD_TO_LDS(g0, g1, g2, g3) \
  __builtin_amdgcn_tensor_load_to_lds((g0), (g1), (g2), (g3), 0)
#endif

__device__ __forceinline__ float gelu_exact(float v) {
  return 0.5f * v * (1.0f + erff(v * 0.70710678118654752f));
}

__device__ __forceinline__ v8f wmma_f16(v16h a, v16h b, v8f c) {
  return __builtin_amdgcn_wmma_f32_16x16x32_f16(false, a, false, b, (short)0, c, false, false);
}

// ---- WMMA fragment-order addressing ----
__device__ __forceinline__ int afrag_idx(int r, int c) {
  return (((c >> 5) * 32) + r + ((c & 16) ? 16 : 0)) * 16 + (c & 15);
}
__device__ __forceinline__ int bfrag_idx(int k, int m, int Mtiles) {
  int lane = (m & 15) + ((k & 16) ? 16 : 0);
  return (((k >> 5) * Mtiles + (m >> 4)) * 32 + lane) * 16 + (k & 15);
}

__device__ __forceinline__ v16h load_a_frag(const _Float16* base, int kk) {
  int lane = threadIdx.x & 31;
  return *((const v16h*)(base + (((kk >> 5) * 32) + lane) * 16));
}
__device__ __forceinline__ v16h load_b_frag(const _Float16* Wf, int Mtiles, int kk, int ct) {
  int lane = threadIdx.x & 31;
  return *((const v16h*)(Wf + (((kk >> 5) * Mtiles + ct) * 32 + lane) * 16));
}

// ------------------------- utility kernels -------------------------

__global__ void k_zero(float* p, long n) {
  long i = (long)blockIdx.x * blockDim.x + threadIdx.x;
  if (i < n) p[i] = 0.0f;
}

__global__ void k_shuffle_w(const float* s, _Float16* d, int K, int M) {
  int i = blockIdx.x * blockDim.x + threadIdx.x;
  if (i >= K * M) return;
  int k = i / M, m = i % M;
  d[bfrag_idx(k, m, M >> 4)] = (_Float16)s[i];
}

__global__ void k_shuffle_pad_aa4(const float* s, _Float16* d) {
  int i = blockIdx.x * blockDim.x + threadIdx.x;
  if (i >= HH * 32) return;
  int k = i / 32, m = i % 32;
  float v = (m < XDIM) ? s[k * XDIM + m] : 0.0f;
  d[bfrag_idx(k, m, 2)] = (_Float16)v;
}

__global__ void k_degree(const int* dst, float* deg) {
  int e = blockIdx.x * blockDim.x + threadIdx.x;
  if (e < N_EDGES) atomicAdd(&deg[dst[e]], 1.0f);
}

__global__ void k_invdeg(float* deg) {
  int i = blockIdx.x * blockDim.x + threadIdx.x;
  if (i < N_NODES) deg[i] = 1.0f / fmaxf(deg[i], 1.0f);
}

__global__ void k_aggregate(const int* srcI, const int* dstI, const float* x, float* agg) {
  int e = blockIdx.x * 8 + (threadIdx.x >> 5);
  if (e >= N_EDGES) return;
  int lane = threadIdx.x & 31;
  long s = srcI[e], t = dstI[e];
  const float* xs = x + s * DD;
  float* at = agg + t * DD;
  int c = lane * 2;
  atomicAdd(&at[c], xs[c]);
  atomicAdd(&at[c + 1], xs[c + 1]);
}

__global__ void k_rows_to_frag(const float* src, const float* invdeg, _Float16* dst) {
  int i4 = blockIdx.x * blockDim.x + threadIdx.x;
  if (i4 >= N_NODES * 16) return;
  long row = i4 >> 4;
  int c0 = (i4 & 15) * 4;
  float sc = invdeg ? invdeg[row] : 1.0f;
  const float4 v = *(const float4*)(src + row * DD + c0);
  _Float16* base = dst + (row >> 4) * (16 * DD);
  int r = (int)(row & 15);
  base[afrag_idx(r, c0 + 0)] = (_Float16)(v.x * sc);
  base[afrag_idx(r, c0 + 1)] = (_Float16)(v.y * sc);
  base[afrag_idx(r, c0 + 2)] = (_Float16)(v.z * sc);
  base[afrag_idx(r, c0 + 3)] = (_Float16)(v.w * sc);
}

// ------------------------- SAGE layer -------------------------
__global__ __launch_bounds__(128) void k_sage(const _Float16* xF, const _Float16* aggF,
                                              const _Float16* WlF, const _Float16* WrF,
                                              const float* bl, float* zcat, float* stats, int layer) {
  int wave = threadIdx.x >> 5;
  int lane = threadIdx.x & 31;
  int rlane = lane & 15;
  long row0 = (long)blockIdx.x * 64 + wave * 16;
  const _Float16* xb = xF + (row0 >> 4) * (16 * DD);
  const _Float16* ab = aggF + (row0 >> 4) * (16 * DD);
  v8f acc[4] = {};
#pragma unroll
  for (int kk = 0; kk < DD; kk += 32) {
    v16h aagg = load_a_frag(ab, kk);
    v16h ax   = load_a_frag(xb, kk);
#pragma unroll
    for (int ct = 0; ct < 4; ++ct) {
      acc[ct] = wmma_f16(aagg, load_b_frag(WlF, 4, kk, ct), acc[ct]);
      acc[ct] = wmma_f16(ax,   load_b_frag(WrF, 4, kk, ct), acc[ct]);
    }
  }
  int rbase = (lane < 16) ? 0 : 8;
#pragma unroll
  for (int ct = 0; ct < 4; ++ct) {
    int col = ct * 16 + rlane;
    float bia = bl[col];
    float s = 0.0f, q = 0.0f;
#pragma unroll
    for (int v = 0; v < 8; ++v) {
      float h = gelu_exact(acc[ct][v] + bia);
      long row = row0 + rbase + v;
      zcat[row * (LAYERS * DD) + layer * DD + col] = h;
      s += h;
      q += h * h;
    }
    atomicAdd(&stats[layer * 2 * DD + col], s);
    atomicAdd(&stats[layer * 2 * DD + DD + col], q);
  }
}

__global__ void k_gnorm(const float* stats, const float* gn_w, const float* gn_b, const float* gn_ms,
                        float* zcat, float* x, _Float16* xF, int layer) {
  long i = (long)blockIdx.x * blockDim.x + threadIdx.x;
  if (i >= (long)N_NODES * DD) return;
  long row = i / DD;
  int col = (int)(i % DD);
  const float invN = 1.0f / (float)N_NODES;
  float mu  = stats[layer * 2 * DD + col] * invN;
  float eh2 = stats[layer * 2 * DD + DD + col] * invN;
  float ms  = gn_ms[layer * DD + col];
  float var = eh2 - 2.0f * ms * mu * mu + ms * ms * mu * mu;
  float h = zcat[row * (LAYERS * DD) + layer * DD + col];
  float xc = h - ms * mu;
  float o = gn_w[layer * DD + col] * xc * rsqrtf(var + 1e-5f) + gn_b[layer * DD + col];
  if (layer > 0) o += x[i];
  zcat[row * (LAYERS * DD) + layer * DD + col] = o;
  x[i] = o;
  xF[(row >> 4) * (16 * DD) + afrag_idx((int)(row & 15), col)] = (_Float16)o;
}

// ------------------------- fused JK MLP -------------------------
__global__ __launch_bounds__(128) void k_mlp(const float* zcat, const float* x_res,
                                             const float* d1w, const float* d1b, const float* d1a,
                                             const _Float16* W1, const float* b1,
                                             const _Float16* W2, const float* b2,
                                             const _Float16* W3, const float* b3,
                                             const float* d2w, const float* d2b, const float* d2a,
                                             float* zout) {
  __shared__ __align__(32) _Float16 bufA[16 * 256];
  __shared__ __align__(32) _Float16 bufB[16 * 256];
  __shared__ __align__(16) float rawz[16 * 192];
  __shared__ float fz[16 * 64];
  __shared__ float rnorm[16];
  long row0 = (long)blockIdx.x * 16;
  int tid = threadIdx.x, wave = tid >> 5, lane = tid & 31, rlane = lane & 15;
  int rbase = (lane < 16) ? 0 : 8;

  // TDM: async-copy the contiguous 16x192 f32 zcat tile into LDS (wave 0 issues)
  if (wave == 0) {
    unsigned long long ga = (unsigned long long)(uintptr_t)(zcat + row0 * 192);
    unsigned ldsa = (unsigned)(uintptr_t)rawz;
    u32x4 g0;
    g0[0] = 1u;                                           // count=1, user descriptor
    g0[1] = ldsa;                                         // lds_addr
    g0[2] = (unsigned)(ga & 0xFFFFFFFFull);               // global_addr[31:0]
    g0[3] = (unsigned)((ga >> 32) & 0x1FFFFFFull) | (2u << 30);  // addr[56:32] | type=2
    i32x8 g1;
    g1[0] = (2 << 16);                                    // data_size=4B
    g1[1] = (int)((192u & 0xFFFFu) << 16);                // tensor_dim0 = 192
    g1[2] = (int)(16u << 16);                             // tensor_dim1 = 16
    g1[3] = (int)(192u << 16);                            // tile_dim0 = 192
    g1[4] = 16;                                           // tile_dim1 = 16
    g1[5] = 192;                                          // tensor_dim0_stride = 192
    g1[6] = 0;
    g1[7] = 0;
    i32x4 gz = {0, 0, 0, 0};
    TDM_LOAD_TO_LDS(g0, g1, gz, gz);
    __builtin_amdgcn_s_wait_tensorcnt(0);
  }
  __syncthreads();

  float a1 = d1a[0];
  for (int e = tid; e < 16 * 192; e += 128) {
    int c = e % 192;
    int r = e / 192;
    float v = rawz[e];
    bufA[afrag_idx(r, c)] = (_Float16)(d1w[c] * tanhf(a1 * v) + d1b[c]);
  }
  __syncthreads();

  { // lin1: 192 -> 256
    v8f acc[4] = {};
    for (int kk = 0; kk < 192; kk += 32) {
      v16h a = load_a_frag(bufA, kk);
#pragma unroll
      for (int ct = 0; ct < 4; ++ct)
        acc[ct] = wmma_f16(a, load_b_frag(W1, 16, kk, wave * 4 + ct), acc[ct]);
    }
#pragma unroll
    for (int ct = 0; ct < 4; ++ct) {
      int col = wave * 64 + ct * 16 + rlane;
      float bia = b1[col];
#pragma unroll
      for (int v = 0; v < 8; ++v)
        bufB[afrag_idx(rbase + v, col)] = (_Float16)gelu_exact(acc[ct][v] + bia);
    }
  }
  __syncthreads();

  { // lin2: 256 -> 256
    v8f acc[4] = {};
    for (int kk = 0; kk < 256; kk += 32) {
      v16h a = load_a_frag(bufB, kk);
#pragma unroll
      for (int ct = 0; ct < 4; ++ct)
        acc[ct] = wmma_f16(a, load_b_frag(W2, 16, kk, wave * 4 + ct), acc[ct]);
    }
#pragma unroll
    for (int ct = 0; ct < 4; ++ct) {
      int col = wave * 64 + ct * 16 + rlane;
      float bia = b2[col];
#pragma unroll
      for (int v = 0; v < 8; ++v)
        bufA[afrag_idx(rbase + v, col)] = (_Float16)gelu_exact(acc[ct][v] + bia);
    }
  }
  __syncthreads();

  { // lin3: 256 -> 64; dyt2 + residual
    v8f acc = {};
    for (int kk = 0; kk < 256; kk += 32)
      acc = wmma_f16(load_a_frag(bufA, kk), load_b_frag(W3, 4, kk, wave), acc);
    int col = wave * 16 + rlane;
    float bia = b3[col];
    float a2 = d2a[0];
#pragma unroll
    for (int v = 0; v < 8; ++v) {
      int r = rbase + v;
      float g = gelu_exact(acc[v] + bia);
      g = d2w[col] * tanhf(a2 * g) + d2b[col];
      g += x_res[(row0 + r) * DD + col];
      fz[r * 64 + col] = g;
    }
  }
  __syncthreads();

  if (tid < 16) {
    float ss = 0.0f;
    for (int c = 0; c < 64; ++c) { float v = fz[tid * 64 + c]; ss += v * v; }
    rnorm[tid] = 1.0f / (sqrtf(ss) + 1e-10f);
  }
  __syncthreads();
  for (int e = tid; e < 16 * 64; e += 128) {
    int r = e >> 6, c = e & 63;
    zout[(row0 + r) * DD + c] = fz[e] * rnorm[r];
  }
}

// ------------------- fused annotation head -------------------
__global__ __launch_bounds__(128) void k_annot(const float* x_res, const float* z,
                                               const float* ad1w, const float* ad1b, const float* ad1a,
                                               const _Float16* A1, const float* ab1,
                                               const _Float16* A2, const float* ab2,
                                               const _Float16* A3, const float* ab3,
                                               const float* ad2w, const float* ad2b, const float* ad2a,
                                               const _Float16* A4p, const float* ab4,
                                               float* out) {
  __shared__ __align__(32) _Float16 bufA[16 * 256];
  __shared__ __align__(32) _Float16 bufB[16 * 256];
  __shared__ float lbuf[16 * 32];
  long row0 = (long)blockIdx.x * 16;
  int tid = threadIdx.x, wave = tid >> 5, lane = tid & 31, rlane = lane & 15;
  int rbase = (lane < 16) ? 0 : 8;

  float al = ad1a[0];
  for (int e = tid; e < 16 * 128; e += 128) {
    int r = e >> 7, c = e & 127;
    float v = (c < 64) ? x_res[(row0 + r) * DD + c] : z[(row0 + r) * DD + (c - 64)];
    bufA[afrag_idx(r, c)] = (_Float16)(ad1w[c] * tanhf(al * v) + ad1b[c]);
  }
  __syncthreads();

  { // aa1: 128 -> 256
    v8f acc[4] = {};
    for (int kk = 0; kk < 128; kk += 32) {
      v16h a = load_a_frag(bufA, kk);
#pragma unroll
      for (int ct = 0; ct < 4; ++ct)
        acc[ct] = wmma_f16(a, load_b_frag(A1, 16, kk, wave * 4 + ct), acc[ct]);
    }
#pragma unroll
    for (int ct = 0; ct < 4; ++ct) {
      int col = wave * 64 + ct * 16 + rlane;
      float bia = ab1[col];
#pragma unroll
      for (int v = 0; v < 8; ++v)
        bufB[afrag_idx(rbase + v, col)] = (_Float16)gelu_exact(acc[ct][v] + bia);
    }
  }
  __syncthreads();

  { // aa2: 256 -> 256
    v8f acc[4] = {};
    for (int kk = 0; kk < 256; kk += 32) {
      v16h a = load_a_frag(bufB, kk);
#pragma unroll
      for (int ct = 0; ct < 4; ++ct)
        acc[ct] = wmma_f16(a, load_b_frag(A2, 16, kk, wave * 4 + ct), acc[ct]);
    }
#pragma unroll
    for (int ct = 0; ct < 4; ++ct) {
      int col = wave * 64 + ct * 16 + rlane;
      float bia = ab2[col];
#pragma unroll
      for (int v = 0; v < 8; ++v)
        bufA[afrag_idx(rbase + v, col)] = (_Float16)gelu_exact(acc[ct][v] + bia);
    }
  }
  __syncthreads();

  { // aa3: 256 -> 256, gelu then adyt2
    v8f acc[4] = {};
    for (int kk = 0; kk < 256; kk += 32) {
      v16h a = load_a_frag(bufA, kk);
#pragma unroll
      for (int ct = 0; ct < 4; ++ct)
        acc[ct] = wmma_f16(a, load_b_frag(A3, 16, kk, wave * 4 + ct), acc[ct]);
    }
    float a2 = ad2a[0];
#pragma unroll
    for (int ct = 0; ct < 4; ++ct) {
      int col = wave * 64 + ct * 16 + rlane;
      float bia = ab3[col];
      float dw = ad2w[col], db = ad2b[col];
#pragma unroll
      for (int v = 0; v < 8; ++v) {
        float g = gelu_exact(acc[ct][v] + bia);
        bufB[afrag_idx(rbase + v, col)] = (_Float16)(dw * tanhf(a2 * g) + db);
      }
    }
  }
  __syncthreads();

  if (wave < 2) { // aa4: 256 -> 32 (cols >= 20 zero-padded)
    v8f acc = {};
    for (int kk = 0; kk < 256; kk += 32)
      acc = wmma_f16(load_a_frag(bufB, kk), load_b_frag(A4p, 2, kk, wave), acc);
    int col = wave * 16 + rlane;
    float bia = (col < XDIM) ? ab4[col] : 0.0f;
#pragma unroll
    for (int v = 0; v < 8; ++v)
      lbuf[(rbase + v) * 32 + col] = acc[v] + bia;
  }
  __syncthreads();

  if (tid < 16) {
    float m = -1e30f;
    for (int c = 0; c < XDIM; ++c) m = fmaxf(m, lbuf[tid * 32 + c]);
    float s = 0.0f;
    for (int c = 0; c < XDIM; ++c) s += expf(lbuf[tid * 32 + c] - m);
    float lse = logf(s);
    for (int c = 0; c < XDIM; ++c)
      out[(row0 + tid) * XDIM + c] = lbuf[tid * 32 + c] - m - lse;
  }
}

// ------------------------- fused contact head -------------------------
__global__ __launch_bounds__(128) void k_contact(const float* z, const int* ci, const int* cj,
                                                 const _Float16* C1, const float* cb1,
                                                 const _Float16* C2, const float* cb2,
                                                 const float* c3w, const float* c3b,
                                                 float* out) {
  __shared__ __align__(32) _Float16 bufA[16 * 128];
  __shared__ __align__(32) _Float16 bufB[16 * 128];
  __shared__ float psum[16][8];
  long p0 = (long)blockIdx.x * 16;
  int tid = threadIdx.x, wave = tid >> 5, lane = tid & 31, rlane = lane & 15;
  int rbase = (lane < 16) ? 0 : 8;

  for (int e4 = tid; e4 < 16 * 32; e4 += 128) {  // float4 granularity gather
    int r = e4 >> 5;
    int c0 = (e4 & 31) * 4;
    long node = (c0 < 64) ? (long)ci[p0 + r] : (long)cj[p0 + r];
    const float4 v = *(const float4*)(z + node * DD + (c0 & 63));
    bufA[afrag_idx(r, c0 + 0)] = (_Float16)v.x;
    bufA[afrag_idx(r, c0 + 1)] = (_Float16)v.y;
    bufA[afrag_idx(r, c0 + 2)] = (_Float16)v.z;
    bufA[afrag_idx(r, c0 + 3)] = (_Float16)v.w;
  }
  __syncthreads();

  { // c1: 128 -> 128
    v8f acc[2] = {};
    for (int kk = 0; kk < 128; kk += 32) {
      v16h a = load_a_frag(bufA, kk);
#pragma unroll
      for (int ct = 0; ct < 2; ++ct)
        acc[ct] = wmma_f16(a, load_b_frag(C1, 8, kk, wave * 2 + ct), acc[ct]);
    }
#pragma unroll
    for (int ct = 0; ct < 2; ++ct) {
      int col = wave * 32 + ct * 16 + rlane;
      float bia = cb1[col];
#pragma unroll
      for (int v = 0; v < 8; ++v)
        bufB[afrag_idx(rbase + v, col)] = (_Float16)gelu_exact(acc[ct][v] + bia);
    }
  }
  __syncthreads();

  { // c2: 128 -> 128
    v8f acc[2] = {};
    for (int kk = 0; kk < 128; kk += 32) {
      v16h a = load_a_frag(bufB, kk);
#pragma unroll
      for (int ct = 0; ct < 2; ++ct)
        acc[ct] = wmma_f16(a, load_b_frag(C2, 8, kk, wave * 2 + ct), acc[ct]);
    }
#pragma unroll
    for (int ct = 0; ct < 2; ++ct) {
      int col = wave * 32 + ct * 16 + rlane;
      float bia = cb2[col];
#pragma unroll
      for (int v = 0; v < 8; ++v)
        bufA[afrag_idx(rbase + v, col)] = (_Float16)gelu_exact(acc[ct][v] + bia);
    }
  }
  __syncthreads();

  { // c3: parallel 128-wide dot per row: thread = (row, 16-col segment)
    int r = tid >> 3, seg = tid & 7;
    float s = 0.0f;
#pragma unroll
    for (int j = 0; j < 16; ++j) {
      int c = seg * 16 + j;
      s += (float)bufA[afrag_idx(r, c)] * c3w[c];
    }
    psum[r][seg] = s;
  }
  __syncthreads();
  if (tid < 16) {
    float s = c3b[0];
#pragma unroll
    for (int j = 0; j < 8; ++j) s += psum[tid][j];
    out[p0 + tid] = 1.0f / (1.0f + expf(-s));
  }
}

// ------------------------- host launcher -------------------------

extern "C" void kernel_launch(void* const* d_in, const int* in_sizes, int n_in,
                              void* d_out, int out_size, void* d_ws, size_t ws_size,
                              hipStream_t stream) {
  (void)in_sizes; (void)n_in; (void)out_size; (void)ws_size;
  const float* x_res  = (const float*)d_in[0];
  const float* Wl     = (const float*)d_in[1];
  const float* bl     = (const float*)d_in[2];
  const float* Wr     = (const float*)d_in[3];
  const float* gn_w   = (const float*)d_in[4];
  const float* gn_b   = (const float*)d_in[5];
  const float* gn_ms  = (const float*)d_in[6];
  const float* d1w    = (const float*)d_in[7];
  const float* d1b    = (const float*)d_in[8];
  const float* d1a    = (const float*)d_in[9];
  const float* lin1_w = (const float*)d_in[10];
  const float* lin1_b = (const float*)d_in[11];
  const float* lin2_w = (const float*)d_in[12];
  const float* lin2_b = (const float*)d_in[13];
  const float* lin3_w = (const float*)d_in[14];
  const float* lin3_b = (const float*)d_in[15];
  const float* d2w    = (const float*)d_in[16];
  const float* d2b    = (const float*)d_in[17];
  const float* d2a    = (const float*)d_in[18];
  const float* ad1w   = (const float*)d_in[19];
  const float* ad1b   = (const float*)d_in[20];
  const float* ad1a   = (const float*)d_in[21];
  const float* aa1_w  = (const float*)d_in[22];
  const float* aa1_b  = (const float*)d_in[23];
  const float* aa2_w  = (const float*)d_in[24];
  const float* aa2_b  = (const float*)d_in[25];
  const float* aa3_w  = (const float*)d_in[26];
  const float* aa3_b  = (const float*)d_in[27];
  const float* ad2w   = (const float*)d_in[28];
  const float* ad2b   = (const float*)d_in[29];
  const float* ad2a   = (const float*)d_in[30];
  const float* aa4_w  = (const float*)d_in[31];
  const float* aa4_b  = (const float*)d_in[32];
  const float* c1_w   = (const float*)d_in[33];
  const float* c1_b   = (const float*)d_in[34];
  const float* c2_w   = (const float*)d_in[35];
  const float* c2_b   = (const float*)d_in[36];
  const float* c3_w   = (const float*)d_in[37];
  const float* c3_b   = (const float*)d_in[38];
  const int* edge     = (const int*)d_in[39];
  const int* cont     = (const int*)d_in[40];
  float* out = (float*)d_out;

  char* wsp = (char*)d_ws;
  auto carve = [&](size_t bytes) -> char* {
    char* p = wsp;
    wsp += (bytes + 255) & ~(size_t)255;
    return p;
  };
  float* deg   = (float*)carve((size_t)N_NODES * 4);
  float* x     = (float*)carve((size_t)N_NODES * DD * 4);
  float* agg   = (float*)carve((size_t)N_NODES * DD * 4);
  float* zcat  = (float*)carve((size_t)N_NODES * LAYERS * DD * 4);
  float* zemb  = (float*)carve((size_t)N_NODES * DD * 4);
  float* stats = (float*)carve((size_t)LAYERS * 2 * DD * 4);
  _Float16* xF   = (_Float16*)carve((size_t)N_NODES * DD * 2);
  _Float16* aggF = (_Float16*)carve((size_t)N_NODES * DD * 2);
  _Float16* WlF = (_Float16*)carve((size_t)LAYERS * DD * DD * 2);
  _Float16* WrF = (_Float16*)carve((size_t)LAYERS * DD * DD * 2);
  _Float16* W1F = (_Float16*)carve((size_t)192 * HH * 2);
  _Float16* W2F = (_Float16*)carve((size_t)HH * HH * 2);
  _Float16* W3F = (_Float16*)carve((size_t)HH * DD * 2);
  _Float16* A1F = (_Float16*)carve((size_t)128 * HH * 2);
  _Float16* A2F = (_Float16*)carve((size_t)HH * HH * 2);
  _Float16* A3F = (_Float16*)carve((size_t)HH * HH * 2);
  _Float16* A4P = (_Float16*)carve((size_t)HH * 32 * 2);
  _Float16* C1F = (_Float16*)carve((size_t)CHN * CHN * 2);
  _Float16* C2F = (_Float16*)carve((size_t)CHN * CHN * 2);

  const int T = 256;
  auto g = [](long n, int t) { return (unsigned)((n + t - 1) / t); };

  for (int l = 0; l < LAYERS; ++l) {
    k_shuffle_w<<<g(DD * DD, T), T, 0, stream>>>(Wl + (size_t)l * DD * DD, WlF + (size_t)l * DD * DD, DD, DD);
    k_shuffle_w<<<g(DD * DD, T), T, 0, stream>>>(Wr + (size_t)l * DD * DD, WrF + (size_t)l * DD * DD, DD, DD);
  }
  k_shuffle_w<<<g(192 * HH, T), T, 0, stream>>>(lin1_w, W1F, 192, HH);
  k_shuffle_w<<<g(HH * HH, T), T, 0, stream>>>(lin2_w, W2F, HH, HH);
  k_shuffle_w<<<g(HH * DD, T), T, 0, stream>>>(lin3_w, W3F, HH, DD);
  k_shuffle_w<<<g(128 * HH, T), T, 0, stream>>>(aa1_w, A1F, 128, HH);
  k_shuffle_w<<<g(HH * HH, T), T, 0, stream>>>(aa2_w, A2F, HH, HH);
  k_shuffle_w<<<g(HH * HH, T), T, 0, stream>>>(aa3_w, A3F, HH, HH);
  k_shuffle_pad_aa4<<<g(HH * 32, T), T, 0, stream>>>(aa4_w, A4P);
  k_shuffle_w<<<g(CHN * CHN, T), T, 0, stream>>>(c1_w, C1F, CHN, CHN);
  k_shuffle_w<<<g(CHN * CHN, T), T, 0, stream>>>(c2_w, C2F, CHN, CHN);

  k_zero<<<g(N_NODES, T), T, 0, stream>>>(deg, N_NODES);
  k_zero<<<g(LAYERS * 2 * DD, T), T, 0, stream>>>(stats, LAYERS * 2 * DD);
  hipMemcpyAsync(x, x_res, (size_t)N_NODES * DD * 4, hipMemcpyDeviceToDevice, stream);
  k_degree<<<g(N_EDGES, T), T, 0, stream>>>(edge + N_EDGES, deg);
  k_invdeg<<<g(N_NODES, T), T, 0, stream>>>(deg);
  k_rows_to_frag<<<g((long)N_NODES * 16, T), T, 0, stream>>>(x_res, nullptr, xF);

  for (int l = 0; l < LAYERS; ++l) {
    k_zero<<<g((long)N_NODES * DD, T), T, 0, stream>>>(agg, (long)N_NODES * DD);
    k_aggregate<<<g(N_EDGES, 8), 256, 0, stream>>>(edge, edge + N_EDGES, x, agg);
    k_rows_to_frag<<<g((long)N_NODES * 16, T), T, 0, stream>>>(agg, deg, aggF);
    k_sage<<<N_NODES / 64, 128, 0, stream>>>(xF, aggF, WlF + (size_t)l * DD * DD,
                                             WrF + (size_t)l * DD * DD, bl + l * DD,
                                             zcat, stats, l);
    k_gnorm<<<g((long)N_NODES * DD, T), T, 0, stream>>>(stats, gn_w, gn_b, gn_ms, zcat, x, xF, l);
  }

  k_mlp<<<N_NODES / 16, 128, 0, stream>>>(zcat, x_res, d1w, d1b, d1a,
                                          W1F, lin1_b, W2F, lin2_b, W3F, lin3_b,
                                          d2w, d2b, d2a, zemb);

  k_annot<<<N_NODES / 16, 128, 0, stream>>>(x_res, zemb, ad1w, ad1b, ad1a,
                                            A1F, aa1_b, A2F, aa2_b, A3F, aa3_b,
                                            ad2w, ad2b, ad2a, A4P, aa4_b, out);

  k_contact<<<N_PAIRS / 16, 128, 0, stream>>>(zemb, cont, cont + N_PAIRS,
                                              C1F, c1_b, C2F, c2_b, c3_w, c3_b,
                                              out + (size_t)N_NODES * XDIM);
}
